// GNNStack_49263274885139
// MI455X (gfx1250) — matrix operand order repaired and checked by reading, not compiled, not run on hardware
//
#include <hip/hip_runtime.h>
#include <hip/hip_bf16.h>

typedef __attribute__((ext_vector_type(16))) _Float16 v16h;
typedef __attribute__((ext_vector_type(8)))  float    v8f;

#define N_NODES 262144
#define N_EDGES 4194304
#define F_INC   128
#define G_GRAPHS 4096
#define NPG     64
#define KSEL    30
#define CATC    97

// ---------------------------------------------------------------------------
// WMMA GEMM: Out[M,COLS] = A[M,K] @ B[K,COLS]  (fp32 in/out, fp16 compute).
// K, COLS compile-time -> K-loop fully unrolls into static v_wmma chains.
// Each wave owns a 16x32 output tile (2 accumulators) so the A fragment is
// loaded once and reused for both column tiles (A-traffic halved; the GEMMs
// are bandwidth-bound so data movement, not math, is the limit).
// BT=false: B is W[K,COLS] row-major (GCN weights).
// BT=true : B is Wt[COLS,K] row-major (fc1_w [128,352]); fragment loads are
//           then contiguous 16-float runs per lane.
// Fragment layouts (wave32): A and B both use contiguous k = kb+16g+e.
// ---------------------------------------------------------------------------
template <int K, int COLS, bool BT>
__global__ __launch_bounds__(256) void gemm_wmma_f16(
    const float* __restrict__ A, int lda,
    const float* __restrict__ W,
    float* __restrict__ Out, int M) {
  constexpr int CP = COLS / 32;              // 32-wide column pairs per row tile
  int wave = (int)((blockIdx.x * blockDim.x + threadIdx.x) >> 5);
  int lane = threadIdx.x & 31;
  int tiles = (M >> 4) * CP;
  if (wave >= tiles) return;                 // wave-uniform: EXEC stays all-1s
  int rowBase = (wave / CP) << 4;
  int colBase = (wave % CP) << 5;
  int r = lane & 15;                         // A row / B,C col within tile
  int g = lane >> 4;                         // lane-group (K-half select)

  v8f acc0 = {}, acc1 = {};
#pragma unroll
  for (int kb = 0; kb < K; kb += 32) {
    // A 16x32 f16 frag: elem e (0..15) -> K=kb+16g+e
    const float* ap = A + (size_t)(rowBase + r) * lda + kb;
    v16h a, b0, b1;
#pragma unroll
    for (int e = 0; e < 8; ++e)  a[e]     = (_Float16)ap[16 * g + e];
#pragma unroll
    for (int e = 0; e < 8; ++e)  a[8 + e] = (_Float16)ap[16 * g + 8 + e];

    if constexpr (BT) {
      // B[k,n] = Wt[n*K + k]; lane n = colBase+r (+16), elem e -> k = kb+16g+e
      const float* w0 = W + (size_t)(colBase + r) * K + kb + 16 * g;
      const float* w1 = w0 + (size_t)16 * K;
#pragma unroll
      for (int e = 0; e < 16; ++e) b0[e] = (_Float16)w0[e];
#pragma unroll
      for (int e = 0; e < 16; ++e) b1[e] = (_Float16)w1[e];
    } else {
      const float* wp = W + (size_t)(kb + 16 * g) * COLS + (colBase + r);
#pragma unroll
      for (int e = 0; e < 16; ++e) b0[e] = (_Float16)wp[(size_t)e * COLS];
#pragma unroll
      for (int e = 0; e < 16; ++e) b1[e] = (_Float16)wp[(size_t)e * COLS + 16];
    }

    acc0 = __builtin_amdgcn_wmma_f32_16x16x32_f16(false, a, false, b0,
                                                  (short)0, acc0, false, false);
    acc1 = __builtin_amdgcn_wmma_f32_16x16x32_f16(false, a, false, b1,
                                                  (short)0, acc1, false, false);
  }
  // C/D layout: VGPR q -> row = rowBase + q + 8g ; col = colBase + r (+16)
  float* op = Out + (size_t)(rowBase + 8 * g) * COLS + (colBase + r);
#pragma unroll
  for (int q = 0; q < 8; ++q) op[(size_t)q * COLS] = acc0[q];
#pragma unroll
  for (int q = 0; q < 8; ++q) op[(size_t)q * COLS + 16] = acc1[q];
}

// ---------------------------------------------------------------------------
// Degree / normalization helpers
// ---------------------------------------------------------------------------
__global__ void init_const_f(float* p, int n, float v) {
  int i = blockIdx.x * blockDim.x + threadIdx.x;
  if (i < n) p[i] = v;
}
__global__ void deg_accum(const int* __restrict__ dst, float* __restrict__ deg) {
  int e = blockIdx.x * blockDim.x + threadIdx.x;
  if (e < N_EDGES) atomicAdd(&deg[dst[e]], 1.0f);
}
__global__ void deg_to_dis(float* p, int n) {
  int i = blockIdx.x * blockDim.x + threadIdx.x;
  if (i < n) p[i] = rsqrtf(p[i]);            // deg >= 1 (self loop) always
}

// ---------------------------------------------------------------------------
// Edge scatter-add, 32-channel layers: one thread per (edge, 4-ch chunk).
// Indices + norm loaded once per chunk; hw gather is a single b128 load.
// agg/hw are 32 MB each -> L2-resident on the 192 MB L2, so these atomics
// never touch HBM in steady state.
// ---------------------------------------------------------------------------
__global__ void edge_agg4(const float* __restrict__ hw, const float* __restrict__ dis,
                          const int* __restrict__ src, const int* __restrict__ dst,
                          float* __restrict__ agg) {
  long long idx = (long long)blockIdx.x * blockDim.x + threadIdx.x;
  if (idx >= (long long)N_EDGES * 8) return;
  int e = (int)(idx >> 3), q = (int)(idx & 7);
  int s = src[e], d = dst[e];
  float nrm = dis[s] * dis[d];
  const float4 h = *(const float4*)(hw + (size_t)s * 32 + q * 4);
  float* ap = agg + (size_t)d * 32 + q * 4;
  atomicAdd(ap + 0, h.x * nrm);
  atomicAdd(ap + 1, h.y * nrm);
  atomicAdd(ap + 2, h.z * nrm);
  atomicAdd(ap + 3, h.w * nrm);
}

// single-channel variant (layer 3)
__global__ void edge_agg1(const float* __restrict__ hw, const float* __restrict__ dis,
                          const int* __restrict__ src, const int* __restrict__ dst,
                          float* __restrict__ agg) {
  int e = blockIdx.x * blockDim.x + threadIdx.x;
  if (e >= N_EDGES) return;
  int s = src[e], d = dst[e];
  atomicAdd(&agg[d], hw[s] * dis[s] * dis[d]);
}

// self-loop + bias + tanh, write into the concat buffer column block
__global__ void finalize_tanh(const float* __restrict__ agg, const float* __restrict__ hw,
                              const float* __restrict__ dis, const float* __restrict__ bias,
                              float* __restrict__ hcat, int C, int colOff) {
  int idx = blockIdx.x * blockDim.x + threadIdx.x;
  if (idx >= N_NODES * C) return;
  int i = idx / C, c = idx % C;
  float di = dis[i];
  float v = agg[idx] + hw[idx] * di * di + bias[c];
  hcat[(size_t)i * CATC + colOff + c] = tanhf(v);
}

// layer-3 projection 32 -> 1 (reads hcat cols 64..95)
__global__ void dot32(const float* __restrict__ hcat, const float* __restrict__ W3,
                      float* __restrict__ hw) {
  int i = blockIdx.x * blockDim.x + threadIdx.x;
  if (i >= N_NODES) return;
  const float* hp = hcat + (size_t)i * CATC + 64;
  float acc = 0.0f;
#pragma unroll
  for (int k = 0; k < 32; ++k) acc += hp[k] * W3[k];
  hw[i] = acc;
}

// ---------------------------------------------------------------------------
// SortAggregation(k=30): per graph, rank 64 nodes by channel 96 descending
// (stable ties -> lower index first, matching argsort(-v)); copy top-30 rows.
// ---------------------------------------------------------------------------
__global__ __launch_bounds__(NPG) void sort_topk(const float* __restrict__ hcat,
                                                 float* __restrict__ top) {
  __shared__ float v[NPG];
  int g = blockIdx.x, t = threadIdx.x;
  int node = g * NPG + t;
  float vi = hcat[(size_t)node * CATC + (CATC - 1)];
  v[t] = vi;
  __syncthreads();
  int rank = 0;
  for (int j = 0; j < NPG; ++j) {
    float vj = v[j];
    rank += (vj > vi) || (vj == vi && j < t);
  }
  if (rank < KSEL) {
    const float* sp = hcat + (size_t)node * CATC;
    float* dp = top + ((size_t)g * KSEL + rank) * CATC;
    for (int c = 0; c < CATC; ++c) dp[c] = sp[c];
  }
}

// conv5 (kernel=stride=97, per-slot dot) + ReLU + MaxPool1d(2,2)
__global__ void conv5_pool(const float* __restrict__ top, const float* __restrict__ w5,
                           const float* __restrict__ b5, float* __restrict__ p5) {
  int idx = blockIdx.x * blockDim.x + threadIdx.x;
  if (idx >= G_GRAPHS * 16 * 15) return;
  int t = idx % 15, o = (idx / 15) % 16, g = idx / (15 * 16);
  const float* wo = w5 + o * CATC;
  float m = 0.0f;  // relu happens before max; max(relu,relu) >= 0
  for (int s = 2 * t; s <= 2 * t + 1; ++s) {
    const float* tp = top + ((size_t)g * KSEL + s) * CATC;
    float acc = b5[o];
    for (int c = 0; c < CATC; ++c) acc += tp[c] * wo[c];
    m = fmaxf(m, acc);
  }
  p5[idx] = m;   // [G,16,15]
}

// conv6: 1D conv k=5 VALID over [G,16,15] -> [G,32,11], ReLU, write flat [G,352]
__global__ void conv6_relu(const float* __restrict__ p5, const float* __restrict__ w6,
                           const float* __restrict__ b6, float* __restrict__ flat) {
  int idx = blockIdx.x * blockDim.x + threadIdx.x;
  if (idx >= G_GRAPHS * 32 * 11) return;
  int t = idx % 11, o = (idx / 11) % 32, g = idx / (11 * 32);
  const float* pg = p5 + (size_t)g * 16 * 15;
  const float* wo = w6 + o * 16 * 5;
  float acc = b6[o];
  for (int i = 0; i < 16; ++i)
#pragma unroll
    for (int dk = 0; dk < 5; ++dk)
      acc += pg[i * 15 + t + dk] * wo[i * 5 + dk];
  flat[(size_t)g * 352 + o * 11 + t] = fmaxf(acc, 0.0f);
}

// fc1 output (pre-bias, from WMMA) -> +bias, ReLU, store out[G,128];
// then fc2 -> log_softmax -> logout[G,2]
__global__ __launch_bounds__(128) void fc2_head(const float* __restrict__ fc1out,
    const float* __restrict__ f1b, const float* __restrict__ f2w,
    const float* __restrict__ f2b, float* __restrict__ out,
    float* __restrict__ logout) {
  __shared__ float s[128];
  __shared__ float sl[2];
  int g = blockIdx.x, j = threadIdx.x;
  float acc = fmaxf(fc1out[(size_t)g * 128 + j] + f1b[j], 0.0f);
  out[(size_t)g * 128 + j] = acc;
  s[j] = acc;
  __syncthreads();
  if (j < 2) {
    const float* w2 = f2w + j * 128;
    float l = f2b[j];
    for (int k = 0; k < 128; ++k) l += s[k] * w2[k];
    sl[j] = l;
  }
  __syncthreads();
  if (j < 2) {
    float m = fmaxf(sl[0], sl[1]);
    float lse = m + logf(expf(sl[0] - m) + expf(sl[1] - m));
    logout[(size_t)g * 2 + j] = sl[j] - lse;
  }
}

// ---------------------------------------------------------------------------
extern "C" void kernel_launch(void* const* d_in, const int* in_sizes, int n_in,
                              void* d_out, int out_size, void* d_ws, size_t ws_size,
                              hipStream_t stream) {
  const float* x   = (const float*)d_in[0];
  const float* W0  = (const float*)d_in[1];  const float* b0 = (const float*)d_in[2];
  const float* W1  = (const float*)d_in[3];  const float* b1 = (const float*)d_in[4];
  const float* W2  = (const float*)d_in[5];  const float* b2 = (const float*)d_in[6];
  const float* W3  = (const float*)d_in[7];  const float* b3 = (const float*)d_in[8];
  const float* c5w = (const float*)d_in[9];  const float* c5b = (const float*)d_in[10];
  const float* c6w = (const float*)d_in[11]; const float* c6b = (const float*)d_in[12];
  const float* f1w = (const float*)d_in[13]; const float* f1b = (const float*)d_in[14];
  const float* f2w = (const float*)d_in[15]; const float* f2b = (const float*)d_in[16];
  const int*   src = (const int*)d_in[17];
  const int*   dst = (const int*)d_in[18];

  float* out_main = (float*)d_out;                     // [G,128]
  float* out_logp = out_main + (size_t)G_GRAPHS * 128; // [G,2]

  // workspace carve-up (~229 MB)
  float* ws     = (float*)d_ws;
  float* dis    = ws;                                      // N
  float* hw     = dis    + (size_t)N_NODES;                // N*32
  float* agg    = hw     + (size_t)N_NODES * 32;           // N*32
  float* hcat   = agg    + (size_t)N_NODES * 32;           // N*97
  float* top    = hcat   + (size_t)N_NODES * CATC;         // G*30*97
  float* p5     = top    + (size_t)G_GRAPHS * KSEL * CATC; // G*16*15
  float* flat   = p5     + (size_t)G_GRAPHS * 16 * 15;     // G*352
  float* fc1out = flat   + (size_t)G_GRAPHS * 352;         // G*128

  const int T = 256;
  auto blk = [](long long n, int t) { return (unsigned)((n + t - 1) / t); };

  // --- degree -> dis = rsqrt(deg) (self loops give deg >= 1) ---
  init_const_f<<<blk(N_NODES, T), T, 0, stream>>>(dis, N_NODES, 1.0f);
  deg_accum<<<blk(N_EDGES, T), T, 0, stream>>>(dst, dis);
  deg_to_dis<<<blk(N_NODES, T), T, 0, stream>>>(dis, N_NODES);

  // one wave per 16x32 tile -> (N/16) waves for the GCN GEMMs
  const unsigned gcnGemmBlocks = blk((long long)(N_NODES / 16) * 32, T);

  // --- GCN layer 0: x[N,128] @ W0 -> hcat cols 0..31 ---
  gemm_wmma_f16<128, 32, false><<<gcnGemmBlocks, T, 0, stream>>>(x, F_INC, W0, hw, N_NODES);
  init_const_f<<<blk((long long)N_NODES * 32, T), T, 0, stream>>>(agg, N_NODES * 32, 0.0f);
  edge_agg4<<<blk((long long)N_EDGES * 8, T), T, 0, stream>>>(hw, dis, src, dst, agg);
  finalize_tanh<<<blk((long long)N_NODES * 32, T), T, 0, stream>>>(agg, hw, dis, b0, hcat, 32, 0);

  // --- GCN layer 1: hcat[:,0:32] @ W1 -> cols 32..63 ---
  gemm_wmma_f16<32, 32, false><<<gcnGemmBlocks, T, 0, stream>>>(hcat + 0, CATC, W1, hw, N_NODES);
  init_const_f<<<blk((long long)N_NODES * 32, T), T, 0, stream>>>(agg, N_NODES * 32, 0.0f);
  edge_agg4<<<blk((long long)N_EDGES * 8, T), T, 0, stream>>>(hw, dis, src, dst, agg);
  finalize_tanh<<<blk((long long)N_NODES * 32, T), T, 0, stream>>>(agg, hw, dis, b1, hcat, 32, 32);

  // --- GCN layer 2: hcat[:,32:64] @ W2 -> cols 64..95 ---
  gemm_wmma_f16<32, 32, false><<<gcnGemmBlocks, T, 0, stream>>>(hcat + 32, CATC, W2, hw, N_NODES);
  init_const_f<<<blk((long long)N_NODES * 32, T), T, 0, stream>>>(agg, N_NODES * 32, 0.0f);
  edge_agg4<<<blk((long long)N_EDGES * 8, T), T, 0, stream>>>(hw, dis, src, dst, agg);
  finalize_tanh<<<blk((long long)N_NODES * 32, T), T, 0, stream>>>(agg, hw, dis, b2, hcat, 32, 64);

  // --- GCN layer 3: hcat[:,64:96] @ W3[32,1] -> col 96 ---
  dot32<<<blk(N_NODES, T), T, 0, stream>>>(hcat, W3, hw);
  init_const_f<<<blk(N_NODES, T), T, 0, stream>>>(agg, N_NODES, 0.0f);
  edge_agg1<<<blk(N_EDGES, T), T, 0, stream>>>(hw, dis, src, dst, agg);
  finalize_tanh<<<blk(N_NODES, T), T, 0, stream>>>(agg, hw, dis, b3, hcat, 1, 96);

  // --- SortAggregation(k=30) ---
  sort_topk<<<G_GRAPHS, NPG, 0, stream>>>(hcat, top);

  // --- conv5 + relu + maxpool2 ---
  conv5_pool<<<blk((long long)G_GRAPHS * 16 * 15, T), T, 0, stream>>>(top, c5w, c5b, p5);

  // --- conv6 + relu -> flat [G,352] ---
  conv6_relu<<<blk((long long)G_GRAPHS * 32 * 11, T), T, 0, stream>>>(p5, c6w, c6b, flat);

  // --- fc1 via WMMA: flat[G,352] @ fc1_w^T[352,128] -> fc1out (pre-bias) ---
  gemm_wmma_f16<352, 128, true><<<blk((long long)(G_GRAPHS / 16) * 4 * 32, T), T, 0, stream>>>(
      flat, 352, f1w, fc1out, G_GRAPHS);

  // --- +bias, relu -> out ; fc2 -> log_softmax ---
  fc2_head<<<G_GRAPHS, 128, 0, stream>>>(fc1out, f1b, f2w, f2b, out_main, out_logp);
}
